// Attention_70025146794457
// MI455X (gfx1250) — compile-verified, matrix operand order checked by reading, not run on hardware
//
#include <hip/hip_runtime.h>
#include <hip/hip_bf16.h>
#include <cstdint>

// ---------------------------------------------------------------- constants
#define BATCH 2
#define SEQ   2048
#define DIM   2048
#define HEADS 16
#define HDIM  128            // DIM / HEADS
#define LOG2E 1.44269504088896340736f

typedef __attribute__((ext_vector_type(16))) __bf16   bf16x16;
typedef __attribute__((ext_vector_type(8)))  float    f32x8;
typedef __attribute__((ext_vector_type(4)))  uint32_t u32x4;
typedef __attribute__((ext_vector_type(8)))  int32_t  i32x8;
typedef __attribute__((ext_vector_type(4)))  int32_t  i32x4;

union Frag16 { bf16x16 v; uint4 q[2]; __bf16 e[16]; };
union Acc8   { f32x8 v;   float f[8]; };

#if defined(__AMDGCN__) && __has_builtin(__builtin_amdgcn_tensor_load_to_lds)
#define HAVE_TDM 1
#else
#define HAVE_TDM 0
#endif

// Load one 16-element bf16 WMMA A/B fragment for this lane.
// CDNA5 16-bit A/B layout: lane (m|n)=lane&15, half=lane>>4,
// element e -> K = f*32 + (e&7) + 8*half + 16*(e>>3): two contiguous
// 16-byte chunks per lane -> ds_load_b128 / global_load_b128.
__device__ __forceinline__ bf16x16 ld_frag(const __bf16* row, int f, int half) {
    Frag16 u;
    u.q[0] = *(const uint4*)(row + f * 32 + 8 * half);
    u.q[1] = *(const uint4*)(row + f * 32 + 16 + 8 * half);
    return u.v;
}

__device__ __forceinline__ f32x8 wmma_bf16(bf16x16 a, bf16x16 b, f32x8 c) {
    return __builtin_amdgcn_wmma_f32_16x16x32_bf16(
        false, a, false, b, (short)0, c, false, false);
}

__device__ __forceinline__ uint32_t lds32(const void* p) {
    return (uint32_t)(uintptr_t)p;       // LDS aperture: offset lives in addr[31:0]
}

#if HAVE_TDM
// Issue a TDM 2D tile load (global -> LDS) with LDS row padding.
// Tile: tile_d1 rows x tile_d0 elements (2-byte data), row stride = stride_d0.
// pad_int_code: DWORDs before pad = 2<<code; pad_amt_code: pad DWORDs - 1.
__device__ __forceinline__ void tdm_load_2d(uint32_t lds_off, const void* gptr,
                                            uint32_t tile_d0, uint32_t tile_d1,
                                            uint32_t stride_d0,
                                            uint32_t pad_int_code, uint32_t pad_amt_code) {
    const uint64_t ga = (uint64_t)(uintptr_t)gptr;
    u32x4 g0;
    g0[0] = 1u;                                            // count=1 (valid), user mode
    g0[1] = lds_off;                                       // lds_addr
    g0[2] = (uint32_t)(ga & 0xFFFFFFFFu);                  // global_addr[31:0]
    g0[3] = (uint32_t)((ga >> 32) & 0x1FFFFFFu) | (2u << 30); // global_addr[56:32], type=2
    i32x8 g1;
    g1[0] = (int32_t)((1u << 16)                           // data_size = 2 bytes
                    | (1u << 20)                           // pad_enable
                    | (pad_int_code << 22)
                    | (pad_amt_code << 25));
    g1[1] = (int32_t)((tile_d0 & 0xFFFFu) << 16);          // tensor_dim0[15:0] @ bits 48..63
    g1[2] = (int32_t)(((tile_d0 >> 16) & 0xFFFFu)          // tensor_dim0[31:16]
                    | ((tile_d1 & 0xFFFFu) << 16));        // tensor_dim1[15:0]
    g1[3] = (int32_t)(((tile_d1 >> 16) & 0xFFFFu)          // tensor_dim1[31:16]
                    | ((tile_d0 & 0xFFFFu) << 16));        // tile_dim0
    g1[4] = (int32_t)(tile_d1 & 0xFFFFu);                  // tile_dim1, tile_dim2=0 (2D)
    g1[5] = (int32_t)stride_d0;                            // tensor_dim0_stride[31:0]
    g1[6] = 0;                                             // stride hi + dim1_stride lo
    g1[7] = 0;
    const i32x4 z4 = (i32x4)0;
#if __clang_major__ >= 23
    const i32x8 z8 = (i32x8)0;
    __builtin_amdgcn_tensor_load_to_lds(g0, g1, z4, z4, z8, 0);
#else
    __builtin_amdgcn_tensor_load_to_lds(g0, g1, z4, z4, 0);
#endif
}
#endif // HAVE_TDM

// ---------------------------------------------------------------- fp32 -> bf16
__global__ void cvt_f32_bf16(const float4* __restrict__ in, uint2* __restrict__ out, int n4) {
    const int i = blockIdx.x * blockDim.x + threadIdx.x;
    if (i < n4) {
        const float4 f = in[i];
        union { __bf16 h[4]; uint2 u; } p;
        p.h[0] = (__bf16)f.x; p.h[1] = (__bf16)f.y;
        p.h[2] = (__bf16)f.z; p.h[3] = (__bf16)f.w;
        out[i] = p.u;
    }
}

// ---------------------------------------------------------------- GEMM
// C[M,N] = A[M,K] * B[N,K]^T (row-major bf16). Tile 128x128, K-step 32.
// 8 waves arranged 4(M) x 2(N); each wave: 32x64 = 8 v_wmma per K-step.
// K tiles double-buffered in LDS; staged by the Tensor Data Mover (wave 0
// issues tensor_load_to_lds, everyone else just syncs), overlapping the next
// tile's DMA with this tile's WMMAs.
__global__ __launch_bounds__(256)
void gemm_bf16_wmma(const __bf16* __restrict__ A, const __bf16* __restrict__ Bw,
                    void* __restrict__ Cout, int M, int N, int K, int out_f32) {
    __shared__ __bf16 As[2][128][40];   // 32 K + 8 pad (16B per row, TDM-padded)
    __shared__ __bf16 Bs[2][128][40];

    const int tid  = threadIdx.x;
    const int wid  = tid >> 5;
    const int lane = tid & 31;
    const int nl   = lane & 15;
    const int half = lane >> 4;
    const int wave_m = wid >> 1;     // 0..3
    const int wave_n = wid & 1;      // 0..1
    const int m0 = blockIdx.y * 128;
    const int n0 = blockIdx.x * 128;
    const int nK = K >> 5;

    Acc8 acc[2][4];
#pragma unroll
    for (int mt = 0; mt < 2; ++mt)
#pragma unroll
        for (int nt = 0; nt < 4; ++nt)
#pragma unroll
            for (int r = 0; r < 8; ++r) acc[mt][nt].f[r] = 0.0f;

#if HAVE_TDM
    if (wid == 0) {   // prime buffer 0 (row = 32 bf16 = 16 DW -> code 3; pad 4 DW -> code 3)
        tdm_load_2d(lds32(&As[0][0][0]), A + (size_t)m0 * K, 32, 128, (uint32_t)K, 3, 3);
        tdm_load_2d(lds32(&Bs[0][0][0]), Bw + (size_t)n0 * K, 32, 128, (uint32_t)K, 3, 3);
    }
#endif

    for (int kt = 0; kt < nK; ++kt) {
        const int buf = kt & 1;
#if HAVE_TDM
        if (wid == 0) __builtin_amdgcn_s_wait_tensorcnt(0);
        __syncthreads();                       // tile `buf` ready for everyone
        if (wid == 0 && kt + 1 < nK) {         // overlap next tile's DMA with compute
            tdm_load_2d(lds32(&As[buf ^ 1][0][0]), A + (size_t)m0 * K + (kt + 1) * 32,
                        32, 128, (uint32_t)K, 3, 3);
            tdm_load_2d(lds32(&Bs[buf ^ 1][0][0]), Bw + (size_t)n0 * K + (kt + 1) * 32,
                        32, 128, (uint32_t)K, 3, 3);
        }
#else
        __syncthreads();
        {
            const int row = tid >> 1, ch = tid & 1;
            const uint4* ga = (const uint4*)(A + (size_t)(m0 + row) * K + kt * 32 + ch * 16);
            const uint4* gb = (const uint4*)(Bw + (size_t)(n0 + row) * K + kt * 32 + ch * 16);
            uint4 a0 = ga[0], a1 = ga[1], b0 = gb[0], b1 = gb[1];
            *(uint4*)&As[buf][row][ch * 16]     = a0;
            *(uint4*)&As[buf][row][ch * 16 + 8] = a1;
            *(uint4*)&Bs[buf][row][ch * 16]     = b0;
            *(uint4*)&Bs[buf][row][ch * 16 + 8] = b1;
        }
        __syncthreads();
#endif
        bf16x16 af[2];
#pragma unroll
        for (int mt = 0; mt < 2; ++mt)
            af[mt] = ld_frag(&As[buf][wave_m * 32 + mt * 16 + nl][0], 0, half);
#pragma unroll
        for (int nt = 0; nt < 4; ++nt) {
            bf16x16 bfv = ld_frag(&Bs[buf][wave_n * 64 + nt * 16 + nl][0], 0, half);
#pragma unroll
            for (int mt = 0; mt < 2; ++mt)
                acc[mt][nt].v = wmma_bf16(af[mt], bfv, acc[mt][nt].v);
        }
    }

    // epilogue: C row = r + 8*half, col = lane&15 (CDNA5 C/D layout)
#pragma unroll
    for (int mt = 0; mt < 2; ++mt)
#pragma unroll
        for (int nt = 0; nt < 4; ++nt) {
            const int gn = n0 + wave_n * 64 + nt * 16 + nl;
#pragma unroll
            for (int r = 0; r < 8; ++r) {
                const int gm = m0 + wave_m * 32 + mt * 16 + 8 * half + r;
                if (out_f32) ((float*)Cout)[(size_t)gm * N + gn] = acc[mt][nt].f[r];
                else        ((__bf16*)Cout)[(size_t)gm * N + gn] = (__bf16)acc[mt][nt].f[r];
            }
        }
}

// ---------------------------------------------------------------- RoPE + L2 norm
// qkv: [B,S,3D] bf16 -> Qt/Kt/Vt: [B,H,S,HD] bf16. One wave per (b,s,h) row.
__global__ __launch_bounds__(256)
void rope_norm(const __bf16* __restrict__ qkv, const float* __restrict__ fc,
               const float* __restrict__ scale_p,
               __bf16* __restrict__ Qt, __bf16* __restrict__ Kt, __bf16* __restrict__ Vt) {
    const int tid = threadIdx.x, wid = tid >> 5, lane = tid & 31;
    const int g = blockIdx.x * 8 + wid;          // 0 .. B*S*H-1
    const int b = g >> 15;                       // S*H = 32768
    const int rem = g & 32767;
    const int s = rem >> 4;
    const int h = rem & 15;
    const float scale = *scale_p;

    const size_t rowbase = ((size_t)b * SEQ + s) * (3 * DIM) + (size_t)h * HDIM;
    const int d0 = lane * 4;                     // 4 elems = 2 rotary pairs

    float q[4], k[4], v[4];
#pragma unroll
    for (int i = 0; i < 4; ++i) {
        q[i] = (float)qkv[rowbase + d0 + i];
        k[i] = (float)qkv[rowbase + DIM + d0 + i];
        v[i] = (float)qkv[rowbase + 2 * DIM + d0 + i];
    }
    const int i0 = d0 >> 1;
    const float c0 = fc[((size_t)s * 64 + i0) * 2 + 0], s0 = fc[((size_t)s * 64 + i0) * 2 + 1];
    const float c1 = fc[((size_t)s * 64 + i0 + 1) * 2 + 0], s1 = fc[((size_t)s * 64 + i0 + 1) * 2 + 1];

    float qr[4], kr[4];
    qr[0] = q[0] * c0 - q[1] * s0;  qr[1] = q[1] * c0 + q[0] * s0;
    qr[2] = q[2] * c1 - q[3] * s1;  qr[3] = q[3] * c1 + q[2] * s1;
    kr[0] = k[0] * c0 - k[1] * s0;  kr[1] = k[1] * c0 + k[0] * s0;
    kr[2] = k[2] * c1 - k[3] * s1;  kr[3] = k[3] * c1 + k[2] * s1;

    float sq = qr[0]*qr[0] + qr[1]*qr[1] + qr[2]*qr[2] + qr[3]*qr[3];
    float sk = kr[0]*kr[0] + kr[1]*kr[1] + kr[2]*kr[2] + kr[3]*kr[3];
#pragma unroll
    for (int mk = 1; mk < 32; mk <<= 1) {
        sq += __shfl_xor(sq, mk, 32);
        sk += __shfl_xor(sk, mk, 32);
    }
    const float qfac = scale / (sqrtf(sq) + 1e-6f);
    const float kfac = 1.0f  / (sqrtf(sk) + 1e-6f);

    const size_t obase = (((size_t)b * HEADS + h) * SEQ + s) * HDIM + d0;
#pragma unroll
    for (int i = 0; i < 4; ++i) {
        Qt[obase + i] = (__bf16)(qr[i] * qfac);
        Kt[obase + i] = (__bf16)(kr[i] * kfac);
        Vt[obase + i] = (__bf16)v[i];
    }
}

// ---------------------------------------------------------------- flash attention
// One workgroup = (b, h, 128 q-rows). 8 waves x 16 rows. K/V blocks of 64.
// Q lives in registers (4 A-fragments). K block staged by TDM; V stored
// transposed in LDS so PV B-fragments read contiguously. P round-trips
// through per-wave LDS to convert C-layout -> A-layout.
__global__ __launch_bounds__(256)
void flash_attn(const __bf16* __restrict__ Qt, const __bf16* __restrict__ Kt,
                const __bf16* __restrict__ Vt, __bf16* __restrict__ O) {
    __shared__ __bf16 Ks[64][136];        // [kk][d]   (TDM pad: 64 DW + 4 DW)
    __shared__ __bf16 Vst[128][72];       // transposed: [d][kk]
    __shared__ __bf16 Ps[8][16][72];      // per-wave P tile [qrow][kcol]

    const int tid = threadIdx.x, wid = tid >> 5, lane = tid & 31;
    const int nl = lane & 15, half = lane >> 4;
    const int qb = blockIdx.x, h = blockIdx.y, b = blockIdx.z;
    const int q0 = qb * 128;
    const size_t headbase = ((size_t)b * HEADS + h) * SEQ;

    // --- Q fragments (16 rows x 128 K) kept resident
    bf16x16 qf[4];
    {
        const __bf16* qrow = Qt + (headbase + q0 + wid * 16 + nl) * HDIM;
#pragma unroll
        for (int f = 0; f < 4; ++f) qf[f] = ld_frag(qrow, f, half);
    }

    float m_i[8], l_i[8];
    Acc8 accv[8];
#pragma unroll
    for (int r = 0; r < 8; ++r) { m_i[r] = -3.0e38f; l_i[r] = 0.0f; }
#pragma unroll
    for (int nt = 0; nt < 8; ++nt)
#pragma unroll
        for (int r = 0; r < 8; ++r) accv[nt].f[r] = 0.0f;

    const int q_hi = q0 + wid * 16 + 15;     // this wave's last q row
    const int nblk = (qb + 1) * 2;           // causal: K blocks up to q0+128

    for (int kb = 0; kb < nblk; ++kb) {
        const int kstart = kb * 64;
        __syncthreads();                     // previous block's LDS reads done
#if HAVE_TDM
        if (wid == 0)                        // row = 128 bf16 = 64 DW (code 5), pad 4 DW (code 3)
            tdm_load_2d(lds32(&Ks[0][0]), Kt + (headbase + kstart) * HDIM,
                        128, 64, 128, 5, 3);
#endif
        {   // --- stage V transposed (batch loads, then scatter to LDS)
            const int c0 = tid, c1 = tid + 256;        // 512 chunks of 16 bf16
            const int kk0 = c0 >> 3, ch0 = c0 & 7;
            const int kk1 = c1 >> 3, ch1 = c1 & 7;
            const uint4* gv0 = (const uint4*)(Vt + (headbase + kstart + kk0) * HDIM + ch0 * 16);
            const uint4* gv1 = (const uint4*)(Vt + (headbase + kstart + kk1) * HDIM + ch1 * 16);
            Frag16 u0, u1;
            u0.q[0] = gv0[0]; u0.q[1] = gv0[1];
            u1.q[0] = gv1[0]; u1.q[1] = gv1[1];
#pragma unroll
            for (int e = 0; e < 16; ++e) Vst[ch0 * 16 + e][kk0] = u0.e[e];
#pragma unroll
            for (int e = 0; e < 16; ++e) Vst[ch1 * 16 + e][kk1] = u1.e[e];
#if !HAVE_TDM
            const uint4* gk0 = (const uint4*)(Kt + (headbase + kstart + kk0) * HDIM + ch0 * 16);
            const uint4* gk1 = (const uint4*)(Kt + (headbase + kstart + kk1) * HDIM + ch1 * 16);
            uint4 k00 = gk0[0], k01 = gk0[1], k10 = gk1[0], k11 = gk1[1];
            *(uint4*)&Ks[kk0][ch0 * 16]     = k00;
            *(uint4*)&Ks[kk0][ch0 * 16 + 8] = k01;
            *(uint4*)&Ks[kk1][ch1 * 16]     = k10;
            *(uint4*)&Ks[kk1][ch1 * 16 + 8] = k11;
#endif
            if (kb + 1 < nblk) {             // hint next block (global_prefetch_b8)
                __builtin_prefetch(Vt + (headbase + kstart + 64 + (tid & 63)) * HDIM, 0, 0);
                __builtin_prefetch(Kt + (headbase + kstart + 64 + (tid & 63)) * HDIM, 0, 0);
            }
        }
#if HAVE_TDM
        if (wid == 0) __builtin_amdgcn_s_wait_tensorcnt(0);
#endif
        __syncthreads();

        if (kstart <= q_hi) {
            // --- scores: Q(16x128) @ K_blk^T(128x64) -> 4 tiles of 16x16
            Acc8 sc[4];
#pragma unroll
            for (int j = 0; j < 4; ++j) {
#pragma unroll
                for (int r = 0; r < 8; ++r) sc[j].f[r] = 0.0f;
#pragma unroll
                for (int f = 0; f < 4; ++f) {
                    bf16x16 kf = ld_frag(&Ks[j * 16 + nl][0], f, half);
                    sc[j].v = wmma_bf16(qf[f], kf, sc[j].v);
                }
            }
            // --- causal mask + row max
            float rowmax[8];
#pragma unroll
            for (int r = 0; r < 8; ++r) rowmax[r] = -3.0e38f;
#pragma unroll
            for (int j = 0; j < 4; ++j)
#pragma unroll
                for (int r = 0; r < 8; ++r) {
                    const int qrow = q0 + wid * 16 + 8 * half + r;
                    const int kcol = kstart + j * 16 + nl;
                    if (kcol > qrow) sc[j].f[r] = -3.0e38f;
                    rowmax[r] = fmaxf(rowmax[r], sc[j].f[r]);
                }
#pragma unroll
            for (int mk = 1; mk < 16; mk <<= 1)
#pragma unroll
                for (int r = 0; r < 8; ++r)
                    rowmax[r] = fmaxf(rowmax[r], __shfl_xor(rowmax[r], mk, 32));

            float alpha[8], rowsum[8];
#pragma unroll
            for (int r = 0; r < 8; ++r) {
                const float mnew = fmaxf(m_i[r], rowmax[r]);
                alpha[r] = exp2f((m_i[r] - mnew) * LOG2E);
                m_i[r] = mnew;
                rowsum[r] = 0.0f;
            }
            // --- P = exp(S - m), spill to LDS in C-layout
#pragma unroll
            for (int j = 0; j < 4; ++j)
#pragma unroll
                for (int r = 0; r < 8; ++r) {
                    const float p = exp2f((sc[j].f[r] - m_i[r]) * LOG2E);
                    rowsum[r] += p;
                    Ps[wid][8 * half + r][j * 16 + nl] = (__bf16)p;
                }
#pragma unroll
            for (int mk = 1; mk < 16; mk <<= 1)
#pragma unroll
                for (int r = 0; r < 8; ++r)
                    rowsum[r] += __shfl_xor(rowsum[r], mk, 32);
#pragma unroll
            for (int r = 0; r < 8; ++r) l_i[r] = l_i[r] * alpha[r] + rowsum[r];
#pragma unroll
            for (int nt = 0; nt < 8; ++nt)
#pragma unroll
                for (int r = 0; r < 8; ++r) accv[nt].f[r] *= alpha[r];

            // --- reload P as A-fragments (same-wave LDS RAW: drain DScnt)
            asm volatile("s_wait_dscnt 0" ::: "memory");
            bf16x16 pf[2];
#pragma unroll
            for (int f = 0; f < 2; ++f) pf[f] = ld_frag(&Ps[wid][nl][0], f, half);

            // --- O += P(16x64) @ V_blk(64x128)
#pragma unroll
            for (int nt = 0; nt < 8; ++nt)
#pragma unroll
                for (int f = 0; f < 2; ++f) {
                    bf16x16 vf = ld_frag(&Vst[nt * 16 + nl][0], f, half);
                    accv[nt].v = wmma_bf16(pf[f], vf, accv[nt].v);
                }
        }
    }

    // --- normalize and write out [B,S,D] with heads concatenated
#pragma unroll
    for (int r = 0; r < 8; ++r) {
        const float inv = 1.0f / l_i[r];
        const int srow = q0 + wid * 16 + 8 * half + r;
#pragma unroll
        for (int nt = 0; nt < 8; ++nt) {
            const int col = h * HDIM + nt * 16 + nl;
            O[((size_t)b * SEQ + srow) * DIM + col] = (__bf16)(accv[nt].f[r] * inv);
        }
    }
}

// ---------------------------------------------------------------- launch
extern "C" void kernel_launch(void* const* d_in, const int* in_sizes, int n_in,
                              void* d_out, int out_size, void* d_ws, size_t ws_size,
                              hipStream_t stream) {
    (void)in_sizes; (void)n_in; (void)out_size; (void)ws_size;
    const float* x      = (const float*)d_in[0];
    const float* w_qkv  = (const float*)d_in[1];
    const float* w_out  = (const float*)d_in[2];
    const float* scale  = (const float*)d_in[3];
    const float* freqs  = (const float*)d_in[4];
    float* out = (float*)d_out;

    const size_t nX  = (size_t)BATCH * SEQ * DIM;       // 8,388,608
    const size_t nWq = (size_t)3 * DIM * DIM;           // 12,582,912
    const size_t nWo = (size_t)DIM * DIM;               // 4,194,304
    const size_t nHead = (size_t)BATCH * HEADS * SEQ * HDIM;

    char* ws = (char*)d_ws;
    size_t off = 0;
    auto carve = [&](size_t bytes) { void* p = ws + off; off = (off + bytes + 255) & ~(size_t)255; return p; };
    __bf16* x_bf    = (__bf16*)carve(nX * 2);
    __bf16* wqkv_bf = (__bf16*)carve(nWq * 2);
    __bf16* wout_bf = (__bf16*)carve(nWo * 2);
    __bf16* qkv_bf  = (__bf16*)carve((size_t)BATCH * SEQ * 3 * DIM * 2);
    __bf16* q_t     = (__bf16*)carve(nHead * 2);
    __bf16* k_t     = (__bf16*)carve(nHead * 2);
    __bf16* v_t     = (__bf16*)carve(nHead * 2);
    __bf16* attn_bf = (__bf16*)carve(nX * 2);

    // 1) fp32 -> bf16 (vectorized: float4 -> packed 4x bf16)
    cvt_f32_bf16<<<(int)(nX  / 4 / 256), 256, 0, stream>>>((const float4*)x,     (uint2*)x_bf,    (int)(nX  / 4));
    cvt_f32_bf16<<<(int)(nWq / 4 / 256), 256, 0, stream>>>((const float4*)w_qkv, (uint2*)wqkv_bf, (int)(nWq / 4));
    cvt_f32_bf16<<<(int)(nWo / 4 / 256), 256, 0, stream>>>((const float4*)w_out, (uint2*)wout_bf, (int)(nWo / 4));

    // 2) qkv = x @ w_qkv^T : M=4096, N=6144, K=2048 (bf16 out)
    gemm_bf16_wmma<<<dim3(3 * DIM / 128, BATCH * SEQ / 128), 256, 0, stream>>>(
        x_bf, wqkv_bf, qkv_bf, BATCH * SEQ, 3 * DIM, DIM, /*out_f32=*/0);

    // 3) RoPE + L2 norm + transpose to [B,H,S,HD]
    rope_norm<<<BATCH * SEQ * HEADS / 8, 256, 0, stream>>>(
        qkv_bf, freqs, scale, q_t, k_t, v_t);

    // 4) causal flash attention -> attn_bf [B,S,D]
    flash_attn<<<dim3(SEQ / 128, HEADS, BATCH), 256, 0, stream>>>(q_t, k_t, v_t, attn_bf);

    // 5) out = attn @ w_out^T : M=4096, N=2048, K=2048 (fp32 out)
    gemm_bf16_wmma<<<dim3(DIM / 128, BATCH * SEQ / 128), 256, 0, stream>>>(
        attn_bf, wout_bf, out, BATCH * SEQ, DIM, DIM, /*out_f32=*/1);
}
